// VQVAE_7971459301761
// MI455X (gfx1250) — matrix operand order repaired and checked by reading, not compiled, not run on hardware
//
#include <hip/hip_runtime.h>

// ---------------------------------------------------------------------------
// VQ-VAE forward for MI455X (gfx1250, wave32, WMMA bf16, async-to-LDS).
//
// argmin(||f||^2+||e||^2-2 f.e) == argmax(f.e - 0.5||e||^2); straight-through
// forward value is exactly embedding[idx]. Pipeline:
//   K1: flat_bf16[N,64]  = bf16( x @ w_in^T + b_in )     (async LDS stage + WMMA)
//   K2: emb_bf16, wout_bf16, halfnorm[k]=0.5||e_k||^2    (VALU prep)
//   K3: idx[n] = argmax_k( flat.e_k - halfnorm[k] )      (WMMA, 2 M-tiles/wave)
//   K4: out[b,o,hw] = w_out @ emb[idx]^T + b_out         (WMMA, coalesced NCHW)
// ---------------------------------------------------------------------------

typedef __bf16 bf16_t;
typedef __bf16 v16bf __attribute__((ext_vector_type(16)));
typedef __bf16 v8bf  __attribute__((ext_vector_type(8)));
typedef float  v8f   __attribute__((ext_vector_type(8)));
typedef float  v16f  __attribute__((ext_vector_type(16)));

#define NUM_TOKENS 8192
#define CODE_DIM   64
#define ENC_CH     512
#define DEC_CH     512
#define BATCH      16
#define HW         1024                           // 32*32
#define NPOS       (BATCH * HW)                   // 16384
#define PROJ_SIZE  ((size_t)BATCH * DEC_CH * HW)  // 8388608 floats

#define WMMA_BF16(A, B, C) \
  __builtin_amdgcn_wmma_f32_16x16x32_bf16(false, (A), false, (B), (short)0, (C), false, false)

// A-matrix 16x32 bf16 fragment: lanes 0-15 hold K {kb+0..15},
// lanes 16-31 hold K {kb+16..31} of row M = lane&15 (matches load_b_frag).
__device__ __forceinline__ v16bf load_a_frag(const bf16_t* __restrict__ row, int kb, int hi) {
  v8bf p0 = *(const v8bf*)(row + kb + hi * 16);
  v8bf p1 = *(const v8bf*)(row + kb + hi * 16 + 8);
  v16bf f;
#pragma unroll
  for (int i = 0; i < 8; ++i) { f[i] = p0[i]; f[8 + i] = p1[i]; }
  return f;
}

// B-matrix 32x16 bf16 fragment: lane holds column N=lane&15, K = kb+hi*16+e
// -> one contiguous 32B run of a 64-wide row.
__device__ __forceinline__ v16bf load_b_frag(const bf16_t* __restrict__ colrow, int kb, int hi) {
  return *(const v16bf*)(colrow + kb + hi * 16);
}

// --------------------------- K1: encoder 1x1 conv ---------------------------
// z[n,d] = sum_c x[b,c,hw]*w_in[d,c] + b_in[d];  M=16384, K=512, N=64.
// Block = 4 waves (the 4 N-tiles) sharing one 16-position x tile staged in LDS
// via gfx1250 async-to-LDS copies (ASYNCcnt), removing 4x redundant x reads.
__global__ void __launch_bounds__(128) vq_encode_kernel(
    const float* __restrict__ x, const float* __restrict__ w_in,
    const float* __restrict__ b_in, bf16_t* __restrict__ flatbf) {
  __shared__ float xtile[ENC_CH * 16];   // 32 KiB, layout [c][p]
  const int tid = threadIdx.x;
  const int lane = tid & 31;
  const int lo = lane & 15, hi = lane >> 4;
  const int mt = blockIdx.x;             // 0..1023 (position tile, 16 contig hw)
  const int nt = tid >> 5;               // 0..3    (codebook-dim tile)
  const int n0 = mt * 16;
  const int b = n0 >> 10, hw0 = n0 & 1023;
  const float* xbase = x + (size_t)b * ENC_CH * HW + hw0;

  // Stage x[b, 0..511, hw0..hw0+15] into LDS: 2048 b128 async copies.
#pragma unroll
  for (int i = 0; i < 16; ++i) {
    const int q = i * 128 + tid;         // 0..2047
    const int c = q >> 2, pseg = q & 3;  // channel, 4-position segment
    const float* gsrc = xbase + (size_t)c * HW + pseg * 4;
    const unsigned lds_off = (unsigned)(size_t)(const void*)(xtile + c * 16 + pseg * 4);
    asm volatile("global_load_async_to_lds_b128 %0, %1, off"
                 :: "v"(lds_off), "v"(gsrc) : "memory");
  }
  asm volatile("s_wait_asynccnt 0x0" ::: "memory");
  __syncthreads();

  const int d = nt * 16 + lo;            // B column = codebook dim
  const float* wrow = w_in + (size_t)d * ENC_CH;
  v8f acc = {};
#pragma unroll 4
  for (int kb = 0; kb < ENC_CH; kb += 32) {
    v16bf af;
#pragma unroll
    for (int e = 0; e < 16; ++e) {
      const int kk = hi * 16 + e;
      af[e] = (bf16_t)xtile[(kb + kk) * 16 + lo];          // ds_load_b32
    }
    v16f wv = *(const v16f*)(wrow + kb + hi * 16);
    v16bf bfg;
#pragma unroll
    for (int e = 0; e < 16; ++e) bfg[e] = (bf16_t)wv[e];
    acc = WMMA_BF16(af, bfg, acc);
  }
  const float bias = b_in[d];
#pragma unroll
  for (int r = 0; r < 8; ++r) {
    const int m = mt * 16 + r + 8 * hi;
    flatbf[(size_t)m * CODE_DIM + d] = (bf16_t)(acc[r] + bias);
  }
}

// ------------------ K2: bf16 conversion + 0.5*||e||^2 prep ------------------
__global__ void __launch_bounds__(256) vq_prep_kernel(
    const float* __restrict__ emb, const float* __restrict__ w_out,
    bf16_t* __restrict__ embbf, bf16_t* __restrict__ woutbf,
    float* __restrict__ halfnorm) {
  const int row = blockIdx.x * blockDim.x + threadIdx.x;
  if (row < NUM_TOKENS) {
    const float* src = emb + (size_t)row * CODE_DIM;
    bf16_t* dst = embbf + (size_t)row * CODE_DIM;
    float s = 0.f;
#pragma unroll 8
    for (int i = 0; i < CODE_DIM; ++i) { const float v = src[i]; s += v * v; dst[i] = (bf16_t)v; }
    halfnorm[row] = 0.5f * s;
  } else if (row < NUM_TOKENS + DEC_CH) {
    const int r2 = row - NUM_TOKENS;
    const float* src = w_out + (size_t)r2 * CODE_DIM;
    bf16_t* dst = woutbf + (size_t)r2 * CODE_DIM;
#pragma unroll 8
    for (int i = 0; i < CODE_DIM; ++i) dst[i] = (bf16_t)src[i];
  }
}

// -------------------- K3: nearest codebook entry (WMMA) ---------------------
// score[n,k] = flat[n].e_k - 0.5||e_k||^2 ; argmax over k.  M=16384, N=8192,
// K=64. Each wave holds TWO M-tiles of A (4 WMMAs per streamed B tile):
// halves codebook L2 traffic while keeping 512 waves of parallelism.
__global__ void __launch_bounds__(128) vq_argmin_kernel(
    const bf16_t* __restrict__ flatbf, const bf16_t* __restrict__ embbf,
    const float* __restrict__ halfnorm, int* __restrict__ idx_out,
    float* __restrict__ idx_f_out) {
  const int lane = threadIdx.x & 31;
  const int lo = lane & 15, hi = lane >> 4;
  const int mb = blockIdx.x * 4 + (threadIdx.x >> 5);   // 0..511 (32-row block)

  v16bf a0[2], a1[2];
#pragma unroll
  for (int t = 0; t < 2; ++t) {
    const bf16_t* arow = flatbf + (size_t)((mb * 2 + t) * 16 + lo) * CODE_DIM;
    a0[t] = load_a_frag(arow, 0, hi);
    a1[t] = load_a_frag(arow, 32, hi);
  }
  float best[2][8];
  int bidx[2][8];
#pragma unroll
  for (int t = 0; t < 2; ++t)
#pragma unroll
    for (int r = 0; r < 8; ++r) { best[t][r] = -3.0e38f; bidx[t][r] = 0; }

#pragma unroll 2
  for (int ntile = 0; ntile < NUM_TOKENS / 16; ++ntile) {
    const int e = ntile * 16 + lo;                      // codebook entry (B col)
    const bf16_t* brow = embbf + (size_t)e * CODE_DIM;
    const v16bf b0 = load_b_frag(brow, 0, hi);
    const v16bf b1 = load_b_frag(brow, 32, hi);
    if (ntile + 1 < NUM_TOKENS / 16)
      __builtin_prefetch(brow + 16 * CODE_DIM, 0, 0);   // global_prefetch_b8
    const float hn = halfnorm[e];
#pragma unroll
    for (int t = 0; t < 2; ++t) {
      v8f acc = {};
      acc = WMMA_BF16(a0[t], b0, acc);
      acc = WMMA_BF16(a1[t], b1, acc);
#pragma unroll
      for (int r = 0; r < 8; ++r) {
        const float sc = acc[r] - hn;
        if (sc > best[t][r]) { best[t][r] = sc; bidx[t][r] = e; }
      }
    }
  }
  // Reduce across the 16 N-lanes (xor masks 1..8 stay inside each hi-half).
#pragma unroll
  for (int m = 1; m < 16; m <<= 1) {
#pragma unroll
    for (int t = 0; t < 2; ++t)
#pragma unroll
      for (int r = 0; r < 8; ++r) {
        const float ov = __shfl_xor(best[t][r], m, 32);
        const int oi = __shfl_xor(bidx[t][r], m, 32);
        if (ov > best[t][r] || (ov == best[t][r] && oi < bidx[t][r])) {
          best[t][r] = ov; bidx[t][r] = oi;
        }
      }
  }
  if (lo == 0) {
#pragma unroll
    for (int t = 0; t < 2; ++t)
#pragma unroll
      for (int r = 0; r < 8; ++r) {
        const int m = (mb * 2 + t) * 16 + r + 8 * hi;
        idx_out[m] = bidx[t][r];
        idx_f_out[m] = (float)bidx[t][r];   // indices, second tuple output
      }
  }
}

// ------------------- K4: gather + decoder 1x1 conv (WMMA) -------------------
// Transposed orientation: A = w_out (M=out-channel), B = emb[idx[n]] (N=pos).
// C lanes then run along hw -> contiguous 64B NCHW stores (the 33.5 MB store
// is half of total HBM traffic). A + bias loaded once, reused for 16 B tiles.
__global__ void __launch_bounds__(128) vq_decode_kernel(
    const bf16_t* __restrict__ embbf, const bf16_t* __restrict__ woutbf,
    const int* __restrict__ idx, const float* __restrict__ b_out,
    float* __restrict__ out) {
  const int lane = threadIdx.x & 31;
  const int lo = lane & 15, hi = lane >> 4;
  const int w = blockIdx.x * 4 + (threadIdx.x >> 5);    // 0..2047
  const int ot = w & 31;                                // out-channel tile
  const int pchunk = w >> 5;                            // 0..63: 16 pos-tiles

  // Loop-invariant A fragments: w_out rows (M = ot*16 + lo).
  const bf16_t* arow = woutbf + (size_t)(ot * 16 + lo) * CODE_DIM;
  const v16bf a0 = load_a_frag(arow, 0, hi);
  const v16bf a1 = load_a_frag(arow, 32, hi);
  float bias[8];
#pragma unroll
  for (int r = 0; r < 8; ++r) bias[r] = b_out[ot * 16 + r + 8 * hi];

  for (int pt = pchunk * 16; pt < pchunk * 16 + 16; ++pt) {
    const int n = pt * 16 + lo;                         // B column = position
    const bf16_t* brow = embbf + (size_t)idx[n] * CODE_DIM;  // quantized row
    const v16bf b0 = load_b_frag(brow, 0, hi);
    const v16bf b1 = load_b_frag(brow, 32, hi);
    v8f acc = {};
    acc = WMMA_BF16(a0, b0, acc);
    acc = WMMA_BF16(a1, b1, acc);
    const int bb = n >> 10, hw = n & 1023;              // same b for all 16 pos
#pragma unroll
    for (int r = 0; r < 8; ++r) {
      const int o = ot * 16 + r + 8 * hi;
      out[((size_t)bb * DEC_CH + o) * HW + hw] = acc[r] + bias[r];
    }
  }
}

// ---------------------------------------------------------------------------
extern "C" void kernel_launch(void* const* d_in, const int* in_sizes, int n_in,
                              void* d_out, int out_size, void* d_ws, size_t ws_size,
                              hipStream_t stream) {
  const float* x     = (const float*)d_in[0];  // [16,512,32,32]
  const float* w_in  = (const float*)d_in[1];  // [64,512]
  const float* b_in  = (const float*)d_in[2];  // [64]
  const float* emb   = (const float*)d_in[3];  // [8192,64]
  const float* w_out = (const float*)d_in[4];  // [512,64]
  const float* b_out = (const float*)d_in[5];  // [512]
  float* out = (float*)d_out;                  // projected (8388608) + indices (16384)

  char* ws = (char*)d_ws;                      // offsets 4 KiB-aligned
  bf16_t* flatbf   = (bf16_t*)(ws + 0);        // 16384*64*2 = 2 MiB
  bf16_t* embbf    = (bf16_t*)(ws + 2097152);  //  8192*64*2 = 1 MiB
  bf16_t* woutbf   = (bf16_t*)(ws + 3145728);  //   512*64*2 = 64 KiB
  float*  halfnorm = (float*) (ws + 3211264);  //    8192*4  = 32 KiB
  int*    idx      = (int*)   (ws + 3244032);  //   16384*4  = 64 KiB

  // K1: encoder GEMM, 1024 blocks x (4 waves sharing one async-staged x tile)
  vq_encode_kernel<<<dim3(NPOS / 16), dim3(128), 0, stream>>>(x, w_in, b_in, flatbf);
  // K2: bf16 + half-norm prep
  vq_prep_kernel<<<dim3((NUM_TOKENS + DEC_CH + 255) / 256), dim3(256), 0, stream>>>(
      emb, w_out, embbf, woutbf, halfnorm);
  // K3: distance GEMM + argmax, 512 waves x 2 M-tiles each
  vq_argmin_kernel<<<dim3(NPOS / 32 / 4), dim3(128), 0, stream>>>(
      flatbf, embbf, halfnorm, idx, out + PROJ_SIZE);
  // K4: gather + decoder GEMM, 2048 waves (32 o-tiles x 64 position chunks)
  vq_decode_kernel<<<dim3((DEC_CH / 16) * 64 / 4), dim3(128), 0, stream>>>(
      embbf, woutbf, idx, b_out, out);
}
